// ScigptMoeDecoderLayerPP_19456201851518
// MI455X (gfx1250) — matrix operand order repaired and checked by reading, not compiled, not run on hardware
//
#include <hip/hip_runtime.h>
#include <hip/hip_bf16.h>
#include <math.h>

// ---------------------------------------------------------------------------
// Problem constants (match reference)
// ---------------------------------------------------------------------------
constexpr int S_  = 2048;
constexpr int H_  = 1024;
constexpr int NH_ = 16;
constexpr int KVH_= 4;
constexpr int HD_ = 64;
constexpr int E_  = 8;
constexpr int F_  = 2048;
constexpr int T_  = 2048;   // tokens = B*S

typedef __attribute__((ext_vector_type(16))) __bf16 v16bf;
typedef __attribute__((ext_vector_type(8)))  float  v8f;
typedef __attribute__((ext_vector_type(4)))  unsigned int v4u;
typedef __attribute__((ext_vector_type(8)))  int v8i;
typedef __attribute__((ext_vector_type(4)))  int v4i;

union AFrag { unsigned short u[16]; v16bf v; };

__device__ __forceinline__ unsigned short f2bf(float f) {
  union { float f; unsigned u; } a; a.f = f;
  unsigned u = a.u;
  unsigned r = u + 0x7FFFu + ((u >> 16) & 1u);
  return (unsigned short)(r >> 16);
}
__device__ __forceinline__ float bf2f(unsigned short h) {
  union { unsigned u; float f; } a; a.u = ((unsigned)h) << 16;
  return a.f;
}
__device__ __forceinline__ v8f zero8() {
  v8f z = {0.f,0.f,0.f,0.f,0.f,0.f,0.f,0.f};
  return z;
}
__device__ __forceinline__ v8f wmma_bf16(v16bf a, v16bf b, v8f c) {
  return __builtin_amdgcn_wmma_f32_16x16x32_bf16(false, a, false, b,
                                                 (short)0, c, false, false);
}

// ---------------------------------------------------------------------------
// Tensor Data Mover: async 2D tile load global -> LDS (CDNA5 TDM).
// D# per CDNA5 ISA. Tracked with TENSORcnt; in-order per wave,
// so partial waits (s_wait_tensorcnt N) enable double-buffered pipelining.
// ---------------------------------------------------------------------------
__device__ __forceinline__ void tdm_load_2d(const void* gptr, unsigned lds_off,
                                            unsigned tile_x, unsigned tile_rows,
                                            unsigned long long row_stride_elems) {
  unsigned long long ga = (unsigned long long)(uintptr_t)gptr;
  v4u g0;
  g0[0] = 1u;                                            // count = 1 descriptor
  g0[1] = lds_off;                                       // lds_addr (bytes)
  g0[2] = (unsigned)(ga & 0xffffffffu);                  // global_addr[31:0]
  g0[3] = (unsigned)((ga >> 32) & 0x01ffffffu) | (2u << 30); // addr[56:32]|type=2
  const unsigned td0 = 0x40000000u, td1 = 0x40000000u;   // huge tensor dims
  v8i g1;
  g1[0] = (int)(1u << 16);                               // data_size = 1 (2 bytes)
  g1[1] = (int)((td0 & 0xffffu) << 16);                  // tensor_dim0[15:0]
  g1[2] = (int)((td0 >> 16) | ((td1 & 0xffffu) << 16));  // td0[31:16], td1[15:0]
  g1[3] = (int)((td1 >> 16) | (tile_x << 16));           // td1[31:16], tile_dim0
  g1[4] = (int)tile_rows;                                // tile_dim1 (tile_dim2=0)
  g1[5] = (int)(row_stride_elems & 0xffffffffu);         // dim0_stride[31:0]
  g1[6] = (int)((row_stride_elems >> 32) & 0xffffu);     // dim0_stride[47:32]
  g1[7] = 0;
  v4i z4 = {0, 0, 0, 0};
  v8i z8 = {0, 0, 0, 0, 0, 0, 0, 0};
  __builtin_amdgcn_tensor_load_to_lds(g0, g1, z4, z4, z8, 0);
}
__device__ __forceinline__ unsigned lds_off(const void* p) {
  return (unsigned)(uintptr_t)p;   // low 32 bits of generic addr = LDS offset
}

// A fragment: 16x32 bf16 tile, row-major with leading dim lda (elements).
__device__ __forceinline__ v16bf load_a_frag(const unsigned short* a, int lda) {
  int lane = threadIdx.x & 31;
  int row  = lane & 15;
  int kb   = (lane >> 4) * 16;
  AFrag f;
#pragma unroll
  for (int j = 0; j < 8; ++j) {
    f.u[j]     = a[row * lda + kb + j];
    f.u[8 + j] = a[row * lda + kb + 8 + j];
  }
  return f.v;
}

// B fragment from weight-style tile Wt[n][k] (B(k,n) = Wt[n][k]).
__device__ __forceinline__ v16bf load_b_frag(const unsigned short* w, int ldw) {
  int lane = threadIdx.x & 31;
  int col  = lane & 15;
  int kb   = (lane >> 4) * 16;
  AFrag f;
#pragma unroll
  for (int j = 0; j < 16; ++j) f.u[j] = w[col * ldw + kb + j];
  return f.v;
}

// B fragment from a natural K-major 32x16 tile: b[k*ldb + n].
__device__ __forceinline__ v16bf load_b_frag_g(const unsigned short* b, int ldb) {
  int lane = threadIdx.x & 31;
  int col  = lane & 15;
  int kb   = (lane >> 4) * 16;
  AFrag f;
#pragma unroll
  for (int j = 0; j < 16; ++j) f.u[j] = b[(size_t)(kb + j) * ldb + col];
  return f.v;
}

// ---------------------------------------------------------------------------
// fp32 -> bf16 conversion (vectorized)
// ---------------------------------------------------------------------------
__global__ void cvt_kernel(const float* __restrict__ x,
                           unsigned short* __restrict__ y, int n) {
  int n4 = n >> 2;
  for (int i = blockIdx.x * blockDim.x + threadIdx.x; i < n4;
       i += gridDim.x * blockDim.x) {
    float4 v = ((const float4*)x)[i];
    ushort4 o;
    o.x = f2bf(v.x); o.y = f2bf(v.y); o.z = f2bf(v.z); o.w = f2bf(v.w);
    ((ushort4*)y)[i] = o;
  }
}

// ---------------------------------------------------------------------------
// RMSNorm (fp32 in, bf16 out). One block (256 thr) per row of H_=1024.
// ---------------------------------------------------------------------------
__global__ void rmsnorm_kernel(const float* __restrict__ x,
                               const float* __restrict__ w,
                               unsigned short* __restrict__ y) {
  int row = blockIdx.x;
  __shared__ float red[8];
  float s = 0.f;
  for (int c = threadIdx.x; c < H_; c += 256) {
    float v = x[(size_t)row * H_ + c];
    s += v * v;
  }
  for (int m = 16; m >= 1; m >>= 1) s += __shfl_xor(s, m, 32);
  if ((threadIdx.x & 31) == 0) red[threadIdx.x >> 5] = s;
  __syncthreads();
  float tot = 0.f;
#pragma unroll
  for (int i = 0; i < 8; ++i) tot += red[i];
  float inv = rsqrtf(tot / (float)H_ + 1e-5f);
  for (int c = threadIdx.x; c < H_; c += 256)
    y[(size_t)row * H_ + c] = f2bf(x[(size_t)row * H_ + c] * inv * w[c]);
}

// ---------------------------------------------------------------------------
// GEMM: C[M,N](f32) = A[M,K](bf16) * W[N,K]^T(bf16) (+ residual)
// Block tile 32x64, 8 waves. Double-buffered TDM: issue next K-step's tiles,
// then partial-wait (TENSORcnt <= 2) so DMA overlaps WMMA.
// ---------------------------------------------------------------------------
__global__ void gemm_bf16_kernel(const unsigned short* __restrict__ A,
                                 const unsigned short* __restrict__ W,
                                 float* __restrict__ C,
                                 const float* __restrict__ residual,
                                 int M, int N, int K) {
  __shared__ __align__(16) unsigned short sA[2][32][32];
  __shared__ __align__(16) unsigned short sB[2][64][32];
  int bm = blockIdx.y * 32, bn = blockIdx.x * 64;
  int wave = threadIdx.x >> 5;
  int wm = (wave >> 2) * 16;
  int wn = (wave & 3) * 16;
  v8f acc = zero8();
  if (threadIdx.x < 32) {           // prologue: stage k0=0 into buffer 0
    tdm_load_2d(A + (size_t)bm * K, lds_off(&sA[0][0][0]), 32, 32, K);
    tdm_load_2d(W + (size_t)bn * K, lds_off(&sB[0][0][0]), 32, 64, K);
  }
  int buf = 0;
  for (int k0 = 0; k0 < K; k0 += 32, buf ^= 1) {
    if (threadIdx.x < 32) {
      if (k0 + 32 < K) {            // issue next tiles, then partial wait
        tdm_load_2d(A + (size_t)bm * K + k0 + 32, lds_off(&sA[buf ^ 1][0][0]),
                    32, 32, K);
        tdm_load_2d(W + (size_t)bn * K + k0 + 32, lds_off(&sB[buf ^ 1][0][0]),
                    32, 64, K);
        __builtin_amdgcn_s_wait_tensorcnt((short)2);
      } else {
        __builtin_amdgcn_s_wait_tensorcnt((short)0);
      }
    }
    __syncthreads();
    v16bf af = load_a_frag(&sA[buf][wm][0], 32);
    v16bf bf = load_b_frag(&sB[buf][wn][0], 32);
    acc = wmma_bf16(af, bf, acc);
    __syncthreads();
  }
  int lane = threadIdx.x & 31;
  int n = lane & 15, mb = (lane >> 4) * 8;
  if (residual) {
#pragma unroll
    for (int r = 0; r < 8; ++r) {
      int row = bm + wm + mb + r, col = bn + wn + n;
      C[(size_t)row * N + col] = acc[r] + residual[(size_t)row * N + col];
    }
  } else {
#pragma unroll
    for (int r = 0; r < 8; ++r) {
      int row = bm + wm + mb + r, col = bn + wn + n;
      C[(size_t)row * N + col] = acc[r];
    }
  }
}

// ---------------------------------------------------------------------------
// RoPE + repack:
//   qf[S][NH*64] -> qb [NH][S][64]  (rotated, bf16)
//   kf[S][KVH*64]-> ktb[KVH][64][S] (rotated, transposed, bf16)
//   vf[S][KVH*64]-> vb [KVH][S][64] (bf16)
// ---------------------------------------------------------------------------
__global__ void rope_pack_kernel(const float* __restrict__ qf,
                                 const float* __restrict__ kf,
                                 const float* __restrict__ vf,
                                 const long long* __restrict__ pos,
                                 unsigned short* __restrict__ qb,
                                 unsigned short* __restrict__ ktb,
                                 unsigned short* __restrict__ vb) {
  int s = blockIdx.x;
  float p = (float)pos[s];
  for (int i = threadIdx.x; i < NH_ * HD_; i += blockDim.x) {
    int hh = i >> 6, d = i & 63, dr = d & 31;
    float inv = __powf(10000.0f, -((float)(2 * dr)) / 64.0f);
    float ang = p * inv, co = __cosf(ang), sn = __sinf(ang);
    float x = qf[(size_t)s * (NH_ * HD_) + i];
    float other = (d < 32) ? -qf[(size_t)s * (NH_ * HD_) + hh * 64 + d + 32]
                           :  qf[(size_t)s * (NH_ * HD_) + hh * 64 + d - 32];
    qb[((size_t)hh * S_ + s) * 64 + d] = f2bf(x * co + other * sn);
  }
  for (int i = threadIdx.x; i < KVH_ * HD_; i += blockDim.x) {
    int hh = i >> 6, d = i & 63, dr = d & 31;
    float inv = __powf(10000.0f, -((float)(2 * dr)) / 64.0f);
    float ang = p * inv, co = __cosf(ang), sn = __sinf(ang);
    float x = kf[(size_t)s * (KVH_ * HD_) + i];
    float other = (d < 32) ? -kf[(size_t)s * (KVH_ * HD_) + hh * 64 + d + 32]
                           :  kf[(size_t)s * (KVH_ * HD_) + hh * 64 + d - 32];
    ktb[((size_t)hh * 64 + d) * S_ + s] = f2bf(x * co + other * sn);
  }
  for (int i = threadIdx.x; i < KVH_ * HD_; i += blockDim.x) {
    int hh = i >> 6, d = i & 63;
    vb[((size_t)hh * S_ + s) * 64 + d] = f2bf(vf[(size_t)s * (KVH_ * HD_) + i]);
  }
}

// ---------------------------------------------------------------------------
// Flash attention (causal, GQA). 4 waves/block, each wave owns a 16-row q
// tile. K/V chunks double-buffered via per-wave TDM with partial TENSORcnt
// waits (wave-local pipeline, no barriers). P transposed through LDS, synced
// with s_wait_dscnt.
// ---------------------------------------------------------------------------
__global__ void flash_attn_kernel(const unsigned short* __restrict__ qb,
                                  const unsigned short* __restrict__ ktb,
                                  const unsigned short* __restrict__ vb,
                                  unsigned short* __restrict__ ctx) {
  __shared__ __align__(16) unsigned short sQ[4][16][64];
  __shared__ __align__(16) unsigned short sK[4][2][64][32];
  __shared__ __align__(16) unsigned short sV[4][2][32][64];
  __shared__ __align__(16) unsigned short sP[4][16][32];
  int wave = threadIdx.x >> 5;
  int lane = threadIdx.x & 31;
  int h = blockIdx.y;
  int kv = h >> 2;                       // NH/KVH = 4
  int q0 = blockIdx.x * 64 + wave * 16;
  const unsigned short* ktbase = ktb + ((size_t)kv * 64) * S_;
  const unsigned short* vbase  = vb + ((size_t)kv * S_) * 64;

  // stage q tile [16][64] via TDM, keep A frags in registers
  tdm_load_2d(qb + ((size_t)h * S_ + q0) * 64, lds_off(&sQ[wave][0][0]), 64, 16, 64);
  __builtin_amdgcn_s_wait_tensorcnt((short)0);
  v16bf aq0 = load_a_frag(&sQ[wave][0][0], 64);
  v16bf aq1 = load_a_frag(&sQ[wave][0][32], 64);

  float mrow[8], lrow[8];
  v8f acc[4];
#pragma unroll
  for (int r = 0; r < 8; ++r) { mrow[r] = -1e30f; lrow[r] = 0.f; }
#pragma unroll
  for (int dt = 0; dt < 4; ++dt) acc[dt] = zero8();
  int n = lane & 15, mb = (lane >> 4) * 8;

  int kcend = q0 + 16;
  // prologue: stage chunk 0 into buffer 0
  tdm_load_2d(ktbase, lds_off(&sK[wave][0][0][0]), 32, 64, (unsigned long long)S_);
  tdm_load_2d(vbase, lds_off(&sV[wave][0][0][0]), 64, 32, 64ull);

  int buf = 0;
  for (int kc = 0; kc < kcend; kc += 32, buf ^= 1) {
    if (kc + 32 < kcend) {               // issue next K/V chunk, partial wait
      tdm_load_2d(ktbase + kc + 32, lds_off(&sK[wave][buf ^ 1][0][0]),
                  32, 64, (unsigned long long)S_);
      tdm_load_2d(vbase + (size_t)(kc + 32) * 64, lds_off(&sV[wave][buf ^ 1][0][0]),
                  64, 32, 64ull);
      __builtin_amdgcn_s_wait_tensorcnt((short)2);
    } else {
      __builtin_amdgcn_s_wait_tensorcnt((short)0);
    }

    v8f s0 = zero8(), s1 = zero8();
    {
      v16bf b0a = load_b_frag_g(&sK[wave][buf][0][0], 32);   // d 0..31, keys 0..15
      v16bf b0b = load_b_frag_g(&sK[wave][buf][32][0], 32);  // d 32..63
      s0 = wmma_bf16(aq0, b0a, s0);
      s0 = wmma_bf16(aq1, b0b, s0);
      v16bf b1a = load_b_frag_g(&sK[wave][buf][0][16], 32);  // keys 16..31
      v16bf b1b = load_b_frag_g(&sK[wave][buf][32][16], 32);
      s1 = wmma_bf16(aq0, b1a, s1);
      s1 = wmma_bf16(aq1, b1b, s1);
    }
    float sv0[8], sv1[8];
#pragma unroll
    for (int r = 0; r < 8; ++r) {
      int row = q0 + mb + r;
      sv0[r] = ((kc + n)      <= row) ? s0[r] * 0.125f : -1e30f;
      sv1[r] = ((kc + 16 + n) <= row) ? s1[r] * 0.125f : -1e30f;
    }
#pragma unroll
    for (int r = 0; r < 8; ++r) {
      float cm = fmaxf(sv0[r], sv1[r]);
      for (int msk = 8; msk >= 1; msk >>= 1) cm = fmaxf(cm, __shfl_xor(cm, msk, 32));
      float mnew  = fmaxf(mrow[r], cm);
      float alpha = __expf(mrow[r] - mnew);
      float p0 = __expf(sv0[r] - mnew);
      float p1 = __expf(sv1[r] - mnew);
      float psum = p0 + p1;
      for (int msk = 8; msk >= 1; msk >>= 1) psum += __shfl_xor(psum, msk, 32);
      lrow[r] = lrow[r] * alpha + psum;
      mrow[r] = mnew;
#pragma unroll
      for (int dt = 0; dt < 4; ++dt) acc[dt][r] *= alpha;
      sP[wave][mb + r][n]      = f2bf(p0);
      sP[wave][mb + r][16 + n] = f2bf(p1);
    }
    asm volatile("s_wait_dscnt 0" ::: "memory");  // wave-internal LDS transpose sync
    v16bf pf = load_a_frag(&sP[wave][0][0], 32);
#pragma unroll
    for (int dt = 0; dt < 4; ++dt) {
      v16bf bv = load_b_frag_g(&sV[wave][buf][0][dt * 16], 64);
      acc[dt] = wmma_bf16(pf, bv, acc[dt]);
    }
  }
#pragma unroll
  for (int dt = 0; dt < 4; ++dt)
#pragma unroll
    for (int r = 0; r < 8; ++r) {
      int row = q0 + mb + r;
      ctx[(size_t)row * (NH_ * HD_) + h * 64 + dt * 16 + n] =
          f2bf(acc[dt][r] / lrow[r]);
    }
}

// ---------------------------------------------------------------------------
// Router: wave per token. logits -> out, softmax, top-2, expert lists.
// ---------------------------------------------------------------------------
__global__ void zero_cnt_kernel(int* cnt) {
  if (threadIdx.x < E_) cnt[threadIdx.x] = 0;
}

__global__ void router_kernel(const unsigned short* __restrict__ xn2,
                              const unsigned short* __restrict__ gate_bf,
                              float* __restrict__ logits_out,
                              int* __restrict__ cnt, int* __restrict__ list,
                              float* __restrict__ wgt, int* __restrict__ slot) {
  int wave = threadIdx.x >> 5, lane = threadIdx.x & 31;
  int t = blockIdx.x * 8 + wave;
  if (t >= T_) return;
  float lg[E_];
#pragma unroll
  for (int e = 0; e < E_; ++e) {
    float s = 0.f;
    for (int c = lane; c < H_; c += 32)
      s += bf2f(xn2[(size_t)t * H_ + c]) * bf2f(gate_bf[(size_t)e * H_ + c]);
    for (int m = 16; m >= 1; m >>= 1) s += __shfl_xor(s, m, 32);
    lg[e] = s;
  }
  if (lane == 0) {
    float mx = lg[0];
#pragma unroll
    for (int e = 1; e < E_; ++e) mx = fmaxf(mx, lg[e]);
    float p[E_], sum = 0.f;
#pragma unroll
    for (int e = 0; e < E_; ++e) { p[e] = __expf(lg[e] - mx); sum += p[e]; }
    int i0 = 0;
#pragma unroll
    for (int e = 1; e < E_; ++e) if (p[e] > p[i0]) i0 = e;
    int i1 = (i0 == 0) ? 1 : 0;
#pragma unroll
    for (int e = 0; e < E_; ++e) if (e != i0 && p[e] > p[i1]) i1 = e;
    float w0 = p[i0] / (p[i0] + p[i1]);
    float w1 = p[i1] / (p[i0] + p[i1]);
#pragma unroll
    for (int e = 0; e < E_; ++e) logits_out[(size_t)t * E_ + e] = lg[e];
    int pos0 = atomicAdd(&cnt[i0], 1);
    list[i0 * T_ + pos0] = t; wgt[i0 * T_ + pos0] = w0; slot[i0 * T_ + pos0] = 2 * t;
    int pos1 = atomicAdd(&cnt[i1], 1);
    list[i1 * T_ + pos1] = t; wgt[i1 * T_ + pos1] = w1; slot[i1 * T_ + pos1] = 2 * t + 1;
  }
}

// ---------------------------------------------------------------------------
// MoE gate+up (fused): g[slot][F] = silu(x@W1e^T) * (x@W3e^T), token-gathered.
// Weight tiles double-buffered via TDM; gathered activations via 128-bit LDS
// copies (single buffer, fenced by the block barriers).
// ---------------------------------------------------------------------------
__global__ void moe_gateup_kernel(const unsigned short* __restrict__ xn2,
                                  const unsigned short* __restrict__ w1,
                                  const unsigned short* __restrict__ w3,
                                  const int* __restrict__ cnt,
                                  const int* __restrict__ list,
                                  const int* __restrict__ slotArr,
                                  unsigned short* __restrict__ gbuf) {
  __shared__ __align__(16) unsigned short sA[32][32];
  __shared__ __align__(16) unsigned short sB1[2][64][32];
  __shared__ __align__(16) unsigned short sB3[2][64][32];
  __shared__ int sTok[32];
  __shared__ int sSlot[32];
  int e = blockIdx.z;
  int ne = cnt[e];
  int p0 = blockIdx.y * 32;
  if (p0 >= ne) return;
  int f0 = blockIdx.x * 64;
  if (threadIdx.x < 32) {
    int p = p0 + (int)threadIdx.x;
    sTok[threadIdx.x]  = (p < ne) ? list[e * T_ + p] : -1;
    sSlot[threadIdx.x] = (p < ne) ? slotArr[e * T_ + p] : -1;
  }
  __syncthreads();
  int wave = threadIdx.x >> 5;
  int wm = (wave >> 2) * 16, wn = (wave & 3) * 16;
  v8f acc1 = zero8(), acc3 = zero8();
  const unsigned short* W1e = w1 + (size_t)e * F_ * H_ + (size_t)f0 * H_;
  const unsigned short* W3e = w3 + (size_t)e * F_ * H_ + (size_t)f0 * H_;
  if (threadIdx.x < 32) {          // prologue: stage k0=0 into buffer 0
    tdm_load_2d(W1e, lds_off(&sB1[0][0][0]), 32, 64, H_);
    tdm_load_2d(W3e, lds_off(&sB3[0][0][0]), 32, 64, H_);
  }
  int buf = 0;
  for (int k0 = 0; k0 < H_; k0 += 32, buf ^= 1) {
    if (threadIdx.x < 32) {
      if (k0 + 32 < H_) {
        tdm_load_2d(W1e + k0 + 32, lds_off(&sB1[buf ^ 1][0][0]), 32, 64, H_);
        tdm_load_2d(W3e + k0 + 32, lds_off(&sB3[buf ^ 1][0][0]), 32, 64, H_);
      }
    }
    for (int i = threadIdx.x; i < 32 * 4; i += 256) {   // gathered A, 16B chunks
      int r = i >> 2, c4 = i & 3;
      int tok = sTok[r];
      uint4 val = make_uint4(0u, 0u, 0u, 0u);
      if (tok >= 0)
        val = ((const uint4*)(xn2 + (size_t)tok * H_ + k0))[c4];
      ((uint4*)&sA[r][0])[c4] = val;
    }
    if (threadIdx.x < 32) {
      if (k0 + 32 < H_) __builtin_amdgcn_s_wait_tensorcnt((short)2);
      else              __builtin_amdgcn_s_wait_tensorcnt((short)0);
    }
    __syncthreads();
    v16bf af = load_a_frag(&sA[wm][0], 32);
    v16bf b1 = load_b_frag(&sB1[buf][wn][0], 32);
    v16bf b3 = load_b_frag(&sB3[buf][wn][0], 32);
    acc1 = wmma_bf16(af, b1, acc1);
    acc3 = wmma_bf16(af, b3, acc3);
    __syncthreads();
  }
  int lane = threadIdx.x & 31;
  int n = lane & 15, mb = (lane >> 4) * 8;
#pragma unroll
  for (int r = 0; r < 8; ++r) {
    int p = p0 + wm + mb + r;
    if (p < ne) {
      int gs = sSlot[wm + mb + r];
      float h1 = acc1[r], h3 = acc3[r];
      float g = (h1 / (1.0f + __expf(-h1))) * h3;
      gbuf[(size_t)gs * F_ + f0 + wn + n] = f2bf(g);
    }
  }
}

// ---------------------------------------------------------------------------
// MoE down: ybuf[slot][H] = wgt * (g[slot] @ W2e^T)
// ---------------------------------------------------------------------------
__global__ void moe_down_kernel(const unsigned short* __restrict__ gbuf,
                                const unsigned short* __restrict__ w2,
                                const int* __restrict__ cnt,
                                const int* __restrict__ slotArr,
                                const float* __restrict__ wgtArr,
                                float* __restrict__ ybuf) {
  __shared__ __align__(16) unsigned short sA[32][32];
  __shared__ __align__(16) unsigned short sB[2][64][32];
  __shared__ int sSlot[32];
  __shared__ float sW[32];
  int e = blockIdx.z;
  int ne = cnt[e];
  int p0 = blockIdx.y * 32;
  if (p0 >= ne) return;
  int h0 = blockIdx.x * 64;
  if (threadIdx.x < 32) {
    int p = p0 + (int)threadIdx.x;
    sSlot[threadIdx.x] = (p < ne) ? slotArr[e * T_ + p] : -1;
    sW[threadIdx.x]    = (p < ne) ? wgtArr[e * T_ + p] : 0.f;
  }
  __syncthreads();
  int wave = threadIdx.x >> 5;
  int wm = (wave >> 2) * 16, wn = (wave & 3) * 16;
  v8f acc = zero8();
  const unsigned short* W2e = w2 + (size_t)e * H_ * F_ + (size_t)h0 * F_;
  if (threadIdx.x < 32)
    tdm_load_2d(W2e, lds_off(&sB[0][0][0]), 32, 64, F_);
  int buf = 0;
  for (int k0 = 0; k0 < F_; k0 += 32, buf ^= 1) {
    if (threadIdx.x < 32) {
      if (k0 + 32 < F_)
        tdm_load_2d(W2e + k0 + 32, lds_off(&sB[buf ^ 1][0][0]), 32, 64, F_);
    }
    for (int i = threadIdx.x; i < 32 * 4; i += 256) {   // gathered g rows
      int r = i >> 2, c4 = i & 3;
      int gs = sSlot[r];
      uint4 val = make_uint4(0u, 0u, 0u, 0u);
      if (gs >= 0)
        val = ((const uint4*)(gbuf + (size_t)gs * F_ + k0))[c4];
      ((uint4*)&sA[r][0])[c4] = val;
    }
    if (threadIdx.x < 32) {
      if (k0 + 32 < F_) __builtin_amdgcn_s_wait_tensorcnt((short)1);
      else              __builtin_amdgcn_s_wait_tensorcnt((short)0);
    }
    __syncthreads();
    v16bf af = load_a_frag(&sA[wm][0], 32);
    v16bf bf = load_b_frag(&sB[buf][wn][0], 32);
    acc = wmma_bf16(af, bf, acc);
    __syncthreads();
  }
  int lane = threadIdx.x & 31;
  int n = lane & 15, mb = (lane >> 4) * 8;
#pragma unroll
  for (int r = 0; r < 8; ++r) {
    int p = p0 + wm + mb + r;
    if (p < ne) {
      int gs = sSlot[wm + mb + r];
      float w = sW[wm + mb + r];
      ybuf[(size_t)gs * H_ + h0 + wn + n] = acc[r] * w;
    }
  }
}

// out = h2 + y[2t] + y[2t+1]   (deterministic combine, no atomics)
__global__ void combine_kernel(const float* __restrict__ h2,
                               const float* __restrict__ ybuf,
                               float* __restrict__ out) {
  int i = blockIdx.x * blockDim.x + threadIdx.x;
  if (i >= S_ * H_) return;
  int t = i / H_, c = i % H_;
  out[i] = h2[i] + ybuf[(size_t)(2 * t) * H_ + c] + ybuf[(size_t)(2 * t + 1) * H_ + c];
}

// ---------------------------------------------------------------------------
// Host launcher
// ---------------------------------------------------------------------------
extern "C" void kernel_launch(void* const* d_in, const int* in_sizes, int n_in,
                              void* d_out, int out_size, void* d_ws, size_t ws_size,
                              hipStream_t stream) {
  (void)in_sizes; (void)n_in; (void)out_size; (void)ws_size;
  const float*     hidden = (const float*)d_in[0];
  const long long* posids = (const long long*)d_in[1];
  const float*     ln1    = (const float*)d_in[2];
  const float*     ln2    = (const float*)d_in[3];
  const float*     wq     = (const float*)d_in[4];
  const float*     wk     = (const float*)d_in[5];
  const float*     wv     = (const float*)d_in[6];
  const float*     wo     = (const float*)d_in[7];
  const float*     gatew  = (const float*)d_in[8];
  const float*     w1     = (const float*)d_in[9];
  const float*     w2     = (const float*)d_in[10];
  const float*     w3     = (const float*)d_in[11];

  float* out        = (float*)d_out;
  float* logits_out = out + (size_t)S_ * H_;

  char* p = (char*)d_ws;
  auto alloc = [&](size_t bytes) -> char* {
    char* r = p;
    p += (bytes + 255) & ~(size_t)255;
    return r;
  };
  unsigned short* wq_b  = (unsigned short*)alloc((size_t)NH_*HD_*H_ * 2);
  unsigned short* wk_b  = (unsigned short*)alloc((size_t)KVH_*HD_*H_ * 2);
  unsigned short* wv_b  = (unsigned short*)alloc((size_t)KVH_*HD_*H_ * 2);
  unsigned short* wo_b  = (unsigned short*)alloc((size_t)H_*NH_*HD_ * 2);
  unsigned short* gt_b  = (unsigned short*)alloc((size_t)E_*H_ * 2);
  unsigned short* w1_b  = (unsigned short*)alloc((size_t)E_*F_*H_ * 2);
  unsigned short* w2_b  = (unsigned short*)alloc((size_t)E_*H_*F_ * 2);
  unsigned short* w3_b  = (unsigned short*)alloc((size_t)E_*F_*H_ * 2);
  unsigned short* xn1_b = (unsigned short*)alloc((size_t)S_*H_ * 2);
  float*          q_f   = (float*)alloc((size_t)S_*NH_*HD_ * 4);
  float*          k_f   = (float*)alloc((size_t)S_*KVH_*HD_ * 4);
  float*          v_f   = (float*)alloc((size_t)S_*KVH_*HD_ * 4);
  unsigned short* q_b   = (unsigned short*)alloc((size_t)NH_*S_*HD_ * 2);
  unsigned short* kt_b  = (unsigned short*)alloc((size_t)KVH_*HD_*S_ * 2);
  unsigned short* v_b   = (unsigned short*)alloc((size_t)KVH_*S_*HD_ * 2);
  unsigned short* ctx_b = (unsigned short*)alloc((size_t)S_*NH_*HD_ * 2);
  float*          h2    = (float*)alloc((size_t)S_*H_ * 4);
  unsigned short* xn2_b = (unsigned short*)alloc((size_t)S_*H_ * 2);
  int*            cnt   = (int*)alloc(E_ * 4);
  int*            list  = (int*)alloc((size_t)E_*T_ * 4);
  float*          wgt   = (float*)alloc((size_t)E_*T_ * 4);
  int*            slot  = (int*)alloc((size_t)E_*T_ * 4);
  unsigned short* gbuf  = (unsigned short*)alloc((size_t)2*T_*F_ * 2);
  float*          ybuf  = (float*)alloc((size_t)2*T_*H_ * 4);

  // ---- weight conversion fp32 -> bf16 ----
  cvt_kernel<<<1024, 256, 0, stream>>>(wq, wq_b, NH_*HD_*H_);
  cvt_kernel<<<512, 256, 0, stream>>>(wk, wk_b, KVH_*HD_*H_);
  cvt_kernel<<<512, 256, 0, stream>>>(wv, wv_b, KVH_*HD_*H_);
  cvt_kernel<<<1024, 256, 0, stream>>>(wo, wo_b, H_*NH_*HD_);
  cvt_kernel<<<32, 256, 0, stream>>>(gatew, gt_b, E_*H_);
  cvt_kernel<<<2048, 256, 0, stream>>>(w1, w1_b, E_*F_*H_);
  cvt_kernel<<<2048, 256, 0, stream>>>(w2, w2_b, E_*H_*F_);
  cvt_kernel<<<2048, 256, 0, stream>>>(w3, w3_b, E_*F_*H_);

  // ---- attention ----
  rmsnorm_kernel<<<S_, 256, 0, stream>>>(hidden, ln1, xn1_b);
  gemm_bf16_kernel<<<dim3((NH_*HD_)/64, S_/32), 256, 0, stream>>>(
      xn1_b, wq_b, q_f, nullptr, S_, NH_*HD_, H_);
  gemm_bf16_kernel<<<dim3((KVH_*HD_)/64, S_/32), 256, 0, stream>>>(
      xn1_b, wk_b, k_f, nullptr, S_, KVH_*HD_, H_);
  gemm_bf16_kernel<<<dim3((KVH_*HD_)/64, S_/32), 256, 0, stream>>>(
      xn1_b, wv_b, v_f, nullptr, S_, KVH_*HD_, H_);
  rope_pack_kernel<<<S_, 256, 0, stream>>>(q_f, k_f, v_f, posids, q_b, kt_b, v_b);
  flash_attn_kernel<<<dim3(S_/64, NH_), 128, 0, stream>>>(q_b, kt_b, v_b, ctx_b);
  gemm_bf16_kernel<<<dim3(H_/64, S_/32), 256, 0, stream>>>(
      ctx_b, wo_b, h2, hidden, S_, H_, NH_*HD_);

  // ---- MoE ----
  rmsnorm_kernel<<<S_, 256, 0, stream>>>(h2, ln2, xn2_b);
  zero_cnt_kernel<<<1, 32, 0, stream>>>(cnt);
  router_kernel<<<T_/8, 256, 0, stream>>>(xn2_b, gt_b, logits_out,
                                          cnt, list, wgt, slot);
  moe_gateup_kernel<<<dim3(F_/64, (T_+31)/32, E_), 256, 0, stream>>>(
      xn2_b, w1_b, w3_b, cnt, list, slot, gbuf);
  moe_down_kernel<<<dim3(H_/64, (T_+31)/32, E_), 256, 0, stream>>>(
      gbuf, w2_b, cnt, slot, wgt, ybuf);
  combine_kernel<<<(S_*H_+255)/256, 256, 0, stream>>>(h2, ybuf, out);
}
